// MultiSpkCTCTokenBuilder_91130616087168
// MI455X (gfx1250) — matrix operand order repaired and checked by reading, not compiled, not run on hardware
//
#include <hip/hip_runtime.h>

// Problem constants (fixed by setup_inputs)
#define KSPK 2
#define BB   8
#define TT   1024
#define DD   512
#define VV   5000
#define NT   313          // ceil(V/16) column tiles
#define FPB  128          // frames per block (8 waves x 16-row M tiles)

typedef __attribute__((ext_vector_type(16))) __bf16 v16bf;
typedef __attribute__((ext_vector_type(8)))  float  v8f;

__device__ __forceinline__ unsigned short f32_to_bf16(float x){
  unsigned int u = __float_as_uint(x);
  if ((u & 0x7F800000u) != 0x7F800000u)        // not inf/nan: round-to-nearest-even
    u += 0x7FFFu + ((u >> 16) & 1u);
  return (unsigned short)(u >> 16);
}

// Online-softmax statistic merge: (m = running max, s = sum exp(x-m), i = argmax).
// Tie-break: smaller index wins (matches jnp.argmax).
__device__ __forceinline__ void mergeStat(float& m, float& s, int& i,
                                          float m2, float s2, int i2){
  float M  = fmaxf(m, m2);
  float sn = s * __expf(m - M) + s2 * __expf(m2 - M);
  if (m2 > m || (m2 == m && i2 < i)) i = i2;
  m = M; s = sn;
}

// ---------------------------------------------------------------------------
// Kernel 0: convert W [K,D,V] fp32 -> bf16 in WMMA B-fragment-native layout.
// Fragment layout (16-bit B 32x16, contiguous k order used by this code, same as A): lanes 0-15 = columns with K=0..15,
// lanes 16-31 same columns with K=16..31; 16 contiguous K values per lane.
// Wb linear index = ((k*NT + nt)*16 + s)*512 + lane*16 + e,
// with d = s*32 + (lane>=16)*16 + e, v = nt*16 + (lane&15).
// One column tile (16 s-steps) is a contiguous 16 KB span.
// ---------------------------------------------------------------------------
__global__ void convert_w(const float* __restrict__ Wf, unsigned short* __restrict__ Wb){
  int l = blockIdx.x * blockDim.x + threadIdx.x;
  if (l >= KSPK * NT * 16 * 512) return;
  int e    = l & 15;
  int lane = (l >> 4) & 31;
  int s    = (l >> 9) & 15;
  int rem  = l >> 13;
  int nt   = rem % NT;
  int k    = rem / NT;
  int d    = s * 32 + (lane >> 4) * 16 + e;
  int v    = nt * 16 + (lane & 15);
  float x  = (v < VV) ? Wf[((size_t)k * DD + d) * VV + v] : 0.f;
  Wb[l] = f32_to_bf16(x);
}

// ---------------------------------------------------------------------------
// Kernel 1: fused GEMM + online argmax/logsumexp.
// grid = (T/128, K*B), 256 threads (8 waves).  Each wave keeps its own 16x512
// A tile register-resident (128 VGPRs, gathered from global with inline bf16
// conversion).  All waves share each 16 KB B column-tile through LDS, staged
// through registers so global loads for tile nt+1 overlap the 16 WMMAs of
// tile nt.  Two independent accumulator chains (even/odd k-steps) and
// alternating B-fragment buffers keep the XDL pipe fed.  Logits never touch
// memory.
// ---------------------------------------------------------------------------
__global__ void __launch_bounds__(256)
ctc_gemm_argmax(const float* __restrict__ sep, const float* __restrict__ bias,
                const unsigned short* __restrict__ Wb,
                int* __restrict__ pathW, float* __restrict__ pbW){
  __shared__ __align__(16) unsigned short sB[16 * 512];   // one 16 KB B tile

  const int tid  = threadIdx.x;
  const int wave = tid >> 5;
  const int lane = tid & 31;
  const int ln   = lane & 15;
  const int hi   = lane >> 4;
  const int kb   = blockIdx.y;        // k*B + b
  const int k    = kb >> 3;
  const int b    = kb & 7;
  const int t0   = blockIdx.x * FPB;

  // --- A: gather this wave's 16x512 tile into registers as bf16 fragments ---
  // Row M = ln of M-tile `wave`; element e holds k = s*32 + 16*hi + e (contiguous order, same as B).
  union AF { unsigned int w[8]; v16bf v; };
  AF aF[16];
  {
    const float* aRow = sep + ((size_t)(k * BB + b) * TT + t0 + wave * 16 + ln) * DD;
    #pragma unroll
    for (int s = 0; s < 16; ++s)
      #pragma unroll
      for (int p = 0; p < 8; ++p){
        int d = s * 32 + hi * 16 + (p << 1);
        const float2 f = *(const float2*)(aRow + d);
        aF[s].w[p] = ((unsigned int)f32_to_bf16(f.y) << 16) | f32_to_bf16(f.x);
      }
  }

  // --- online argmax/logsumexp state: 8 rows per lane, this lane's columns ---
  float mM[8], sS[8], blk[8]; int iI[8];
  #pragma unroll
  for (int e = 0; e < 8; ++e){ mM[e] = -3.0e38f; sS[e] = 0.f; iI[e] = 0x7fffffff; blk[e] = 0.f; }

  const unsigned short* wbase = Wb + ((size_t)k * NT * 16) * 512;

  // preload stage registers with tile 0 (each thread owns 64 B of the tile)
  uint4 stg[4];
  {
    const uint4* src = (const uint4*)wbase + tid * 4;
    stg[0] = src[0]; stg[1] = src[1]; stg[2] = src[2]; stg[3] = src[3];
  }

  for (int nt = 0; nt < NT; ++nt){
    __syncthreads();                       // sB free to overwrite
    {
      uint4* dst = (uint4*)sB + tid * 4;
      dst[0] = stg[0]; dst[1] = stg[1]; dst[2] = stg[2]; dst[3] = stg[3];
    }
    if (nt + 1 < NT){                      // pipeline: fetch next tile now
      const uint4* src = (const uint4*)(wbase + (size_t)(nt + 1) * (16 * 512)) + tid * 4;
      stg[0] = src[0]; stg[1] = src[1]; stg[2] = src[2]; stg[3] = src[3];
    }
    if (nt + 2 < NT)                       // gfx1250 global_prefetch_b8
      __builtin_prefetch(wbase + (size_t)(nt + 2) * (16 * 512) + tid * 32, 0, 1);
    __syncthreads();                       // sB ready

    const int v = nt * 16 + ln;
    const float bv = (v < VV) ? bias[k * VV + v] : 0.f;

    // Two independent WMMA accumulator chains: even steps start from the bias,
    // odd steps start from zero; summed after the K loop.
    v8f c0, c1;
    #pragma unroll
    for (int e = 0; e < 8; ++e){ c0[e] = bv; c1[e] = 0.f; }

    union BF { uint4 q[2]; v16bf v; };
    BF Bf[2];                              // alternating B-fragment buffers
    {
      const uint4* bp = (const uint4*)(sB + lane * 16);
      Bf[0].q[0] = bp[0]; Bf[0].q[1] = bp[1];
    }

    #pragma unroll
    for (int s = 0; s < 16; ++s){          // K loop: 16 x (K=32) = D=512
      if (s + 1 < 16){                     // load next fragment into other buffer
        const uint4* bp = (const uint4*)(sB + (s + 1) * 512 + lane * 16);
        Bf[(s + 1) & 1].q[0] = bp[0];
        Bf[(s + 1) & 1].q[1] = bp[1];
      }
      if ((s & 1) == 0)
        c0 = __builtin_amdgcn_wmma_f32_16x16x32_bf16(false, aF[s].v, false, Bf[s & 1].v,
                                                     (short)0, c0, false, false);
      else
        c1 = __builtin_amdgcn_wmma_f32_16x16x32_bf16(false, aF[s].v, false, Bf[s & 1].v,
                                                     (short)0, c1, false, false);
    }

    if (v < VV){
      #pragma unroll
      for (int e = 0; e < 8; ++e){
        float x = c0[e] + c1[e];
        if (x > mM[e]){ sS[e] = sS[e] * __expf(mM[e] - x) + 1.f; mM[e] = x; iI[e] = v; }
        else          { sS[e] += __expf(x - mM[e]); }
      }
    }
    if (nt == 0 && ln == 0){               // column v==0 is the blank logit
      #pragma unroll
      for (int e = 0; e < 8; ++e) blk[e] = c0[e] + c1[e];
    }
  }

  // Butterfly merge across the 16 lanes that share each row (different columns).
  #pragma unroll
  for (int off = 1; off < 16; off <<= 1){
    #pragma unroll
    for (int e = 0; e < 8; ++e){
      float m2 = __shfl_xor(mM[e], off, 32);
      float s2 = __shfl_xor(sS[e], off, 32);
      int   i2 = __shfl_xor(iI[e], off, 32);
      mergeStat(mM[e], sS[e], iI[e], m2, s2, i2);
    }
  }

  // ln==0 lanes hold both the merged stats and the blank logits for their rows.
  if (ln == 0){
    #pragma unroll
    for (int e = 0; e < 8; ++e){
      int t = t0 + wave * 16 + hi * 8 + e;
      float pb = __expf(blk[e] - mM[e]) / sS[e];   // exp(log_softmax[blank])
      pathW[kb * TT + t] = iI[e];
      pbW[kb * TT + t]   = pb;
    }
  }
}

// ---------------------------------------------------------------------------
// Kernel 2: per-(k,b) CTC run-length scan -> segment ranges + mask/conf outputs.
// ---------------------------------------------------------------------------
__global__ void seg_scan(const int* __restrict__ pathW, const float* __restrict__ pbW,
                         const unsigned char* __restrict__ enc,
                         int* __restrict__ segA, int* __restrict__ segN,
                         float* __restrict__ pm,
                         float* __restrict__ outMask, float* __restrict__ outConf){
  __shared__ int sNs;
  const int kb = blockIdx.x;
  const int k = kb >> 3, b = kb & 7;

  for (int s = threadIdx.x; s < TT; s += blockDim.x) segN[kb * TT + s] = 0;
  __syncthreads();

  if (threadIdx.x == 0){
    int ns = 0, st = 0, cn = 0, prevTok = -1; float pp = 0.f;
    bool prevValid = false, inRange = true;
    for (int t = 0; t < TT; ++t){
      inRange = inRange && (enc[b * TT + t] != 0);   // prefix-valid region
      int tok = pathW[kb * TT + t];
      bool valid = inRange && (tok != 0);            // BLANK_ID == 0
      if (valid){
        if (!prevValid || tok != prevTok){           // new segment
          if (ns > 0){ segA[kb*TT+ns-1]=st; segN[kb*TT+ns-1]=cn; pm[kb*TT+ns-1]=pp/(float)cn; }
          st = t; cn = 0; pp = 0.f; ns++;
        }
        cn++; pp += pbW[kb * TT + t];
      }
      prevValid = valid; prevTok = tok;
    }
    if (ns > 0){ segA[kb*TT+ns-1]=st; segN[kb*TT+ns-1]=cn; pm[kb*TT+ns-1]=pp/(float)cn; }
    sNs = ns;
  }
  __syncthreads();

  const int ns = sNs;
  for (int s = threadIdx.x; s < TT; s += blockDim.x){
    size_t o = (size_t)b * (KSPK * TT) + k * TT + s;
    bool pad = s >= ns;
    float c = pad ? 0.f : fminf(fmaxf(1.f - pm[kb * TT + s], 0.f), 1.f);
    outMask[o] = pad ? 1.f : 0.f;
    outConf[o] = c;
  }
}

// ---------------------------------------------------------------------------
// Kernel 3: segment-mean features (gather over contiguous frame runs), zero-fill pad.
// grid = (T, K*B), 128 threads, float4 (b128) loads/stores over D=512.
// ---------------------------------------------------------------------------
__global__ void seg_avg(const float* __restrict__ sep, const int* __restrict__ segA,
                        const int* __restrict__ segN, float* __restrict__ outMem){
  const int s  = blockIdx.x;
  const int kb = blockIdx.y;
  const int k = kb >> 3, b = kb & 7;
  const int d = threadIdx.x * 4;

  const int n = segN[kb * TT + s];
  float4 acc = make_float4(0.f, 0.f, 0.f, 0.f);
  if (n > 0){
    const int a0 = segA[kb * TT + s];
    const float inv = 1.f / (float)n;
    for (int f = a0; f < a0 + n; ++f){
      const float4 v = *(const float4*)(sep + (((size_t)(k * BB + b) * TT + f) * DD + d));
      acc.x += v.x; acc.y += v.y; acc.z += v.z; acc.w += v.w;
    }
    acc.x *= inv; acc.y *= inv; acc.z *= inv; acc.w *= inv;
  }
  *(float4*)(outMem + (((size_t)b * (KSPK * TT) + k * TT + s) * DD + d)) = acc;
}

// ---------------------------------------------------------------------------
extern "C" void kernel_launch(void* const* d_in, const int* in_sizes, int n_in,
                              void* d_out, int out_size, void* d_ws, size_t ws_size,
                              hipStream_t stream) {
  (void)in_sizes; (void)n_in; (void)out_size; (void)ws_size;

  const float*         sep  = (const float*)d_in[0];         // [K,B,T,D] f32
  const float*         Wf   = (const float*)d_in[1];         // [K,D,V]   f32
  const float*         bias = (const float*)d_in[2];         // [K,V]     f32
  const unsigned char* enc  = (const unsigned char*)d_in[3]; // [B,T]     bool

  // Workspace layout (~10.6 MB total)
  char* ws = (char*)d_ws;
  unsigned short* Wb = (unsigned short*)ws;                  // bf16 B-frag layout
  size_t off = (size_t)KSPK * NT * 16 * 512 * 2;             // 10,256,384 B
  off = (off + 255) & ~(size_t)255;
  int*   pathW = (int*)  (ws + off); off += (size_t)KSPK * BB * TT * 4;
  float* pbW   = (float*)(ws + off); off += (size_t)KSPK * BB * TT * 4;
  int*   segA  = (int*)  (ws + off); off += (size_t)KSPK * BB * TT * 4;
  int*   segN  = (int*)  (ws + off); off += (size_t)KSPK * BB * TT * 4;
  float* pm    = (float*)(ws + off);

  float* outMem  = (float*)d_out;                                  // [B,K*T,D]
  float* outMask = outMem  + (size_t)BB * KSPK * TT * DD;          // [B,K*T]
  float* outConf = outMask + (size_t)BB * KSPK * TT;               // [B,K*T]

  const int totW = KSPK * NT * 16 * 512;
  convert_w<<<(totW + 255) / 256, 256, 0, stream>>>(Wf, Wb);
  ctc_gemm_argmax<<<dim3(TT / FPB, KSPK * BB), 256, 0, stream>>>(sep, bias, Wb, pathW, pbW);
  seg_scan<<<dim3(KSPK * BB), 256, 0, stream>>>(pathW, pbW, enc, segA, segN, pm, outMask, outConf);
  seg_avg<<<dim3(TT, KSPK * BB), 128, 0, stream>>>(sep, segA, segN, outMem);
}
